// CoNe_35974646071945
// MI455X (gfx1250) — matrix operand order repaired and checked by reading, not compiled, not run on hardware
//
#include <hip/hip_runtime.h>
#include <hip/hip_bf16.h>

// ---------------------------------------------------------------------------
// Problem constants
// ---------------------------------------------------------------------------
#define BB   512       // batch
#define DD   256       // feature dim
#define KK   65536     // queue length
#define CC   1000      // classes
#define M1   1024      // stacked rows for GEMM1 (norm_q ; k_feat)
#define KNN  200
#define T_SUP 0.07f
#define T_DC  0.10f
#define LSM   0.10f    // label smoothing
#define EPSV  1e-8f
#define USTRIDE 1024   // padded N stride for U accumulator

typedef __attribute__((ext_vector_type(16))) __bf16 v16bf;
typedef __attribute__((ext_vector_type(8)))  __bf16 v8bf;
typedef __attribute__((ext_vector_type(8)))  float  v8f;

__device__ __forceinline__ __bf16 f2bf(float f) {
    union { float f; unsigned u; } x; x.f = f;
    unsigned r = (x.u + 0x7FFFu + ((x.u >> 16) & 1u)) >> 16;
    union { unsigned short s; __bf16 b; } y; y.s = (unsigned short)r;
    return y.b;
}

__device__ __forceinline__ unsigned monokey(float v) {
    unsigned u = __float_as_uint(v);
    return (u & 0x80000000u) ? ~u : (u | 0x80000000u);
}

// CDNA5 async global->LDS copy (ASYNCcnt-tracked, no VGPR bounce).
// ldsOff: wave-relative LDS byte offset (flat addr low 32 bits, ISA 10.2).
__device__ __forceinline__ void async_b128(unsigned ldsOff, const void* g) {
    unsigned long long ga = (unsigned long long)g;
    asm volatile("global_load_async_to_lds_b128 %0, %1, off"
                 :: "v"(ldsOff), "v"(ga) : "memory");
}
__device__ __forceinline__ void wait_async4() {
    asm volatile("s_wait_asynccnt 0x4" ::: "memory");
}
__device__ __forceinline__ void wait_async0() {
    asm volatile("s_wait_asynccnt 0x0" ::: "memory");
}

// ---------------------------------------------------------------------------
// 1. init: zero the 3 outputs and the split-K accumulator U
// ---------------------------------------------------------------------------
__global__ void k_init(float* out, float* U) {
    int i = blockIdx.x * blockDim.x + threadIdx.x;
    if (i < 3) out[i] = 0.0f;
    if (i < BB * USTRIDE) U[i] = 0.0f;
}

// ---------------------------------------------------------------------------
// 2. stack norm_q / k_feat and convert to bf16  [1024][256]
// ---------------------------------------------------------------------------
__global__ void k_convA(const float* __restrict__ nq, const float* __restrict__ kf,
                        __bf16* __restrict__ A) {
    int idx = blockIdx.x * blockDim.x + threadIdx.x;           // < M1*DD
    int m = idx >> 8, d = idx & 255;
    float v = (m < BB) ? nq[m * DD + d] : kf[(m - BB) * DD + d];
    A[idx] = f2bf(v);
}

// ---------------------------------------------------------------------------
// 3. transpose queue [D][K] fp32 -> queueT [K][D] bf16 (LDS tiled)
// ---------------------------------------------------------------------------
__global__ void k_transQ(const float* __restrict__ q, __bf16* __restrict__ qt) {
    __shared__ float tile[32][33];
    int k0 = blockIdx.x * 32, d0 = blockIdx.y * 32;
    int tx = threadIdx.x, ty = threadIdx.y;
    tile[ty][tx] = q[(size_t)(d0 + ty) * KK + (k0 + tx)];
    __syncthreads();
    qt[(size_t)(k0 + ty) * DD + (d0 + tx)] = f2bf(tile[tx][ty]);
}

// ---------------------------------------------------------------------------
// 4. queue_label_prob fp32 [C][K] -> bf16 same layout
// ---------------------------------------------------------------------------
__global__ void k_convQLP(const float* __restrict__ p, __bf16* __restrict__ o) {
    size_t n = (size_t)CC * KK;
    for (size_t i = blockIdx.x * (size_t)blockDim.x + threadIdx.x; i < n;
         i += (size_t)gridDim.x * blockDim.x)
        o[i] = f2bf(p[i]);
}

// ---------------------------------------------------------------------------
// WMMA fragment loads (A 16x32 bf16, B 32x16), contiguous k order: lane half h holds k = 16h..16h+15.
// ---------------------------------------------------------------------------
__device__ __forceinline__ v16bf loadAfrag(const __bf16* sA, int mrow, int hi) {
    return *(const v16bf*)(sA + mrow * 32 + 16 * hi);
}
__device__ __forceinline__ v16bf loadBfrag(const __bf16* sB, int nrow, int hi) {
    return *(const v16bf*)(sB + nrow * 32 + 16 * hi);
}

// ---------------------------------------------------------------------------
// 5. GEMM1:  S[1024][65536] = A[1024][256] * queueT[65536][256]^T   (fp32 out)
//    Block 256 threads = 8 waves (2M x 4N); tile 128x128, BK=32.
//    Double-buffered async global->LDS staging (4 async b128 per thread/batch).
//    launch_bounds(256,2): 2 waves/SIMD is the mandatory residency for an
//    8-wave workgroup -> give each wave half the VGPR file (no spills).
// ---------------------------------------------------------------------------
__global__ void __launch_bounds__(256, 2)
k_gemm1(const __bf16* __restrict__ A, const __bf16* __restrict__ Bq,
        float* __restrict__ S) {
    __shared__ __bf16 sA[2][128 * 32];
    __shared__ __bf16 sB[2][128 * 32];
    const int tid = threadIdx.x;
    const int lane = tid & 31, hi = lane >> 4, ml = lane & 15;
    const int wave = tid >> 5, wm = wave >> 2, wn = wave & 3;
    const int m0 = blockIdx.y * 128, n0 = blockIdx.x * 128;

    const int row0 = tid >> 2, col0 = (tid & 3) * 8;
    const __bf16* gA0 = A  + (size_t)(m0 + row0) * DD + col0;
    const __bf16* gA1 = A  + (size_t)(m0 + row0 + 64) * DD + col0;
    const __bf16* gB0 = Bq + (size_t)(n0 + row0) * DD + col0;
    const __bf16* gB1 = Bq + (size_t)(n0 + row0 + 64) * DD + col0;
    const unsigned oA0 = (unsigned)(size_t)&sA[0][row0 * 32 + col0];
    const unsigned oB0 = (unsigned)(size_t)&sB[0][row0 * 32 + col0];

    auto issue = [&](int buf, int kk) {
        const unsigned d = (unsigned)buf * 8192u;
        async_b128(oA0 + d,        gA0 + kk);
        async_b128(oA0 + d + 4096, gA1 + kk);
        async_b128(oB0 + d,        gB0 + kk);
        async_b128(oB0 + d + 4096, gB1 + kk);
    };

    v8f acc[4][2] = {};
    issue(0, 0);
    for (int t = 0; t < DD / 32; ++t) {
        const int buf = t & 1;
        if (t < DD / 32 - 1) { issue(buf ^ 1, (t + 1) * 32); wait_async4(); }
        else                 { wait_async0(); }
        __syncthreads();
        const __bf16* pA = &sA[buf][0];
        const __bf16* pB = &sB[buf][0];
        v16bf af[4], bfr[2];
#pragma unroll
        for (int m = 0; m < 4; ++m) af[m] = loadAfrag(pA, wm * 64 + m * 16 + ml, hi);
#pragma unroll
        for (int n = 0; n < 2; ++n) bfr[n] = loadBfrag(pB, wn * 32 + n * 16 + ml, hi);
#pragma unroll
        for (int m = 0; m < 4; ++m)
#pragma unroll
            for (int n = 0; n < 2; ++n)
                acc[m][n] = __builtin_amdgcn_wmma_f32_16x16x32_bf16(
                    false, af[m], false, bfr[n], (short)0, acc[m][n], false, false);
        __syncthreads();
    }
#pragma unroll
    for (int m = 0; m < 4; ++m)
#pragma unroll
        for (int n = 0; n < 2; ++n) {
            int gm = m0 + wm * 64 + m * 16 + 8 * hi;
            int gn = n0 + wn * 32 + n * 16 + ml;
#pragma unroll
            for (int r = 0; r < 8; ++r)
                S[(size_t)(gm + r) * KK + gn] = acc[m][n][r];
        }
}

// ---------------------------------------------------------------------------
// 6. dc softmax numerator: E[b][k] = bf16(exp((x-max)/T_DC)), rowsum stored.
// ---------------------------------------------------------------------------
__global__ void k_expdc(const float* __restrict__ S, __bf16* __restrict__ E,
                        float* __restrict__ rowsum) {
    const int b = blockIdx.x, tid = threadIdx.x;
    const float* row = S + (size_t)(BB + b) * KK;
    __shared__ float red[256];
    float mx = -3.402823466e38f;
    for (int i = tid; i < KK; i += 256) mx = fmaxf(mx, row[i]);
    red[tid] = mx; __syncthreads();
    for (int s = 128; s > 0; s >>= 1) { if (tid < s) red[tid] = fmaxf(red[tid], red[tid + s]); __syncthreads(); }
    const float m = red[0];
    __syncthreads();
    float sum = 0.0f;
    for (int i = tid; i < KK; i += 256) {
        float e = __expf((row[i] - m) * (1.0f / T_DC));
        E[(size_t)b * KK + i] = f2bf(e);
        sum += e;
    }
    red[tid] = sum; __syncthreads();
    for (int s = 128; s > 0; s >>= 1) { if (tid < s) red[tid] += red[tid + s]; __syncthreads(); }
    if (tid == 0) rowsum[b] = red[0];
}

// ---------------------------------------------------------------------------
// 7. exact top-200 per row via 2-level 11-bit radix-histogram select,
//    then supcon softmax-weighted positive mass.
// ---------------------------------------------------------------------------
__global__ void k_topk(const float* __restrict__ S, const int* __restrict__ qlabel,
                       const int* __restrict__ target, float* __restrict__ out) {
    const int b = blockIdx.x, tid = threadIdx.x;
    const float* row = S + (size_t)b * KK;
    __shared__ unsigned hist[2048];
    __shared__ float red[256], red2[256];
    __shared__ unsigned eqTaken;
    __shared__ int s_b1, s_cnt1, s_b2, s_r;
    __shared__ float s_max;

    for (int i = tid; i < 2048; i += 256) hist[i] = 0u;
    __syncthreads();
    float mx = -3.402823466e38f;
    for (int i = tid; i < KK; i += 256) {
        float v = row[i];
        mx = fmaxf(mx, v);
        atomicAdd(&hist[monokey(v) >> 21], 1u);
    }
    red[tid] = mx; __syncthreads();
    for (int s = 128; s > 0; s >>= 1) { if (tid < s) red[tid] = fmaxf(red[tid], red[tid + s]); __syncthreads(); }
    if (tid == 0) {
        s_max = red[0];
        int cum = 0, bin = 2047;
        for (; bin > 0; --bin) { int c = (int)hist[bin]; if (cum + c >= KNN) break; cum += c; }
        s_b1 = bin; s_cnt1 = cum;
    }
    __syncthreads();
    const int b1 = s_b1, cntAbove = s_cnt1;

    for (int i = tid; i < 2048; i += 256) hist[i] = 0u;
    __syncthreads();
    for (int i = tid; i < KK; i += 256) {
        unsigned k = monokey(row[i]);
        if ((int)(k >> 21) == b1) atomicAdd(&hist[(k >> 10) & 0x7FFu], 1u);
    }
    __syncthreads();
    if (tid == 0) {
        int need = KNN - cntAbove;
        int cum = 0, bin = 2047;
        for (; bin > 0; --bin) { int c = (int)hist[bin]; if (cum + c >= need) break; cum += c; }
        s_b2 = bin; s_r = need - cum; eqTaken = 0u;
    }
    __syncthreads();
    const unsigned thr22 = (((unsigned)b1) << 11) | (unsigned)s_b2;
    const int r = s_r;
    const float mxv = s_max;
    const int tgt = target[b];
    float se = 0.0f, sp = 0.0f;
    for (int i = tid; i < KK; i += 256) {
        float v = row[i];
        unsigned p22 = monokey(v) >> 10;
        bool take = (p22 > thr22);
        if (!take && p22 == thr22) { unsigned o = atomicAdd(&eqTaken, 1u); take = ((int)o < r); }
        if (take) {
            float e = __expf((v - mxv) * (1.0f / T_SUP));
            se += e;
            if (qlabel[i] == tgt) sp += e;
        }
    }
    red[tid] = se; red2[tid] = sp; __syncthreads();
    for (int s = 128; s > 0; s >>= 1) {
        if (tid < s) { red[tid] += red[tid + s]; red2[tid] += red2[tid + s]; }
        __syncthreads();
    }
    if (tid == 0) {
        float gt = red2[0] / red[0];
        if (gt > EPSV) atomicAdd(out + 0, -__logf(gt) * (1.0f / BB));
    }
}

// ---------------------------------------------------------------------------
// 8. GEMM2 (split-K x32): U[512][1000] += E[512][65536] * qlp[1000][65536]^T
//    OOB N rows are clamped to row CC-1 (their columns are never stored),
//    keeping every wave's async batch exactly 4 instructions.
// ---------------------------------------------------------------------------
__global__ void __launch_bounds__(256, 2)
k_gemm2(const __bf16* __restrict__ E, const __bf16* __restrict__ P,
        float* __restrict__ U) {
    __shared__ __bf16 sA[2][128 * 32];
    __shared__ __bf16 sB[2][128 * 32];
    const int tid = threadIdx.x;
    const int lane = tid & 31, hi = lane >> 4, ml = lane & 15;
    const int wave = tid >> 5, wm = wave >> 2, wn = wave & 3;
    const int m0 = blockIdx.y * 128, n0 = blockIdx.x * 128;
    const int k0 = blockIdx.z * 2048;

    const int row0 = tid >> 2, col0 = (tid & 3) * 8;
    const int cb0 = min(n0 + row0, CC - 1);
    const int cb1 = min(n0 + row0 + 64, CC - 1);
    const __bf16* gA0 = E + (size_t)(m0 + row0) * KK + k0 + col0;
    const __bf16* gA1 = E + (size_t)(m0 + row0 + 64) * KK + k0 + col0;
    const __bf16* gB0 = P + (size_t)cb0 * KK + k0 + col0;
    const __bf16* gB1 = P + (size_t)cb1 * KK + k0 + col0;
    const unsigned oA0 = (unsigned)(size_t)&sA[0][row0 * 32 + col0];
    const unsigned oB0 = (unsigned)(size_t)&sB[0][row0 * 32 + col0];

    auto issue = [&](int buf, int kk) {
        const unsigned d = (unsigned)buf * 8192u;
        async_b128(oA0 + d,        gA0 + kk);
        async_b128(oA0 + d + 4096, gA1 + kk);
        async_b128(oB0 + d,        gB0 + kk);
        async_b128(oB0 + d + 4096, gB1 + kk);
    };

    v8f acc[4][2] = {};
    issue(0, 0);
    for (int t = 0; t < 64; ++t) {
        const int buf = t & 1;
        if (t < 63) { issue(buf ^ 1, (t + 1) * 32); wait_async4(); }
        else        { wait_async0(); }
        __syncthreads();
        const __bf16* pA = &sA[buf][0];
        const __bf16* pB = &sB[buf][0];
        v16bf af[4], bfr[2];
#pragma unroll
        for (int m = 0; m < 4; ++m) af[m] = loadAfrag(pA, wm * 64 + m * 16 + ml, hi);
#pragma unroll
        for (int n = 0; n < 2; ++n) bfr[n] = loadBfrag(pB, wn * 32 + n * 16 + ml, hi);
#pragma unroll
        for (int m = 0; m < 4; ++m)
#pragma unroll
            for (int n = 0; n < 2; ++n)
                acc[m][n] = __builtin_amdgcn_wmma_f32_16x16x32_bf16(
                    false, af[m], false, bfr[n], (short)0, acc[m][n], false, false);
        __syncthreads();
    }
#pragma unroll
    for (int m = 0; m < 4; ++m)
#pragma unroll
        for (int n = 0; n < 2; ++n) {
            int gm = m0 + wm * 64 + m * 16 + 8 * hi;
            int gn = n0 + wn * 32 + n * 16 + ml;
            if (gn < CC)
#pragma unroll
                for (int r = 0; r < 8; ++r)
                    atomicAdd(&U[(size_t)(gm + r) * USTRIDE + gn], acc[m][n][r]);
        }
}

// ---------------------------------------------------------------------------
// 9. fc loss (masked label-smoothed CE) + per-row LSE / mask for dc loss
// ---------------------------------------------------------------------------
__global__ void k_fc(const float* __restrict__ ql, const int* __restrict__ target,
                     float* __restrict__ out, float* __restrict__ lse,
                     float* __restrict__ maskf) {
    const int b = blockIdx.x, tid = threadIdx.x;
    const float* x = ql + (size_t)b * CC;
    __shared__ float r1[256], r2[256], r3[256];
    float mx = -3.402823466e38f, mn = 3.402823466e38f, sx = 0.0f;
    for (int i = tid; i < CC; i += 256) { float v = x[i]; mx = fmaxf(mx, v); mn = fminf(mn, v); sx += v; }
    r1[tid] = mx; r2[tid] = mn; r3[tid] = sx; __syncthreads();
    for (int s = 128; s > 0; s >>= 1) {
        if (tid < s) { r1[tid] = fmaxf(r1[tid], r1[tid + s]); r2[tid] = fminf(r2[tid], r2[tid + s]); r3[tid] += r3[tid + s]; }
        __syncthreads();
    }
    const float M = r1[0], MN = r2[0], SX = r3[0];
    __syncthreads();
    float se = 0.0f;
    for (int i = tid; i < CC; i += 256) se += __expf(x[i] - M);
    r1[tid] = se; __syncthreads();
    for (int s = 128; s > 0; s >>= 1) { if (tid < s) r1[tid] += r1[tid + s]; __syncthreads(); }
    if (tid == 0) {
        float L = M + __logf(r1[0]);                // log-sum-exp
        float mk = ((MN - L) > __logf(EPSV)) ? 1.0f : 0.0f;
        float logq_t = x[target[b]] - L;
        float sumlogq = SX - (float)CC * L;
        float fcb = -((LSM / (CC - 1)) * (sumlogq - logq_t) + (1.0f - LSM) * logq_t) * mk;
        atomicAdd(out + 1, fcb * (1.0f / BB));
        lse[b] = L; maskf[b] = mk;
    }
}

// ---------------------------------------------------------------------------
// 10. dc loss: KL(dc_target || softmax(q_logits)); dc_target = U / rowsum
// ---------------------------------------------------------------------------
__global__ void k_dc(const float* __restrict__ U, const float* __restrict__ rowsum,
                     const float* __restrict__ ql, const float* __restrict__ lse,
                     const float* __restrict__ maskf, float* __restrict__ out) {
    const int b = blockIdx.x, tid = threadIdx.x;
    if (maskf[b] == 0.0f) return;
    const float inv = 1.0f / rowsum[b];
    const float L = lse[b];
    __shared__ float red[256];
    float acc = 0.0f;
    for (int c = tid; c < CC; c += 256) {
        float t = U[(size_t)b * USTRIDE + c] * inv;
        if (t > 0.0f) {
            float lq = ql[(size_t)b * CC + c] - L;
            acc += t * (__logf(t) - lq);
        }
    }
    red[tid] = acc; __syncthreads();
    for (int s = 128; s > 0; s >>= 1) { if (tid < s) red[tid] += red[tid + s]; __syncthreads(); }
    if (tid == 0) atomicAdd(out + 2, red[0] * (1.0f / BB));
}

// ---------------------------------------------------------------------------
// launch
// ---------------------------------------------------------------------------
extern "C" void kernel_launch(void* const* d_in, const int* in_sizes, int n_in,
                              void* d_out, int out_size, void* d_ws, size_t ws_size,
                              hipStream_t stream) {
    const float* norm_q  = (const float*)d_in[0];
    const float* q_logits= (const float*)d_in[1];
    const float* k_feat  = (const float*)d_in[2];
    // d_in[3] = logits_k (unused by reference)
    const float* queue   = (const float*)d_in[4];
    const float* qlp     = (const float*)d_in[5];
    const int*   qlabel  = (const int*)d_in[6];
    const int*   target  = (const int*)d_in[7];
    float* out = (float*)d_out;

    char* w = (char*)d_ws;
    size_t off = 0;
    auto take = [&](size_t bytes) { char* p = w + off; off = (off + bytes + 255) & ~(size_t)255; return p; };
    __bf16* bfA   = (__bf16*)take((size_t)M1 * DD * 2);       //   0.5 MB
    __bf16* bfQT  = (__bf16*)take((size_t)KK * DD * 2);       //  32 MB
    __bf16* bfQLP = (__bf16*)take((size_t)CC * KK * 2);       // 125 MB
    float*  S     = (float*) take((size_t)M1 * KK * 4);       // 256 MB
    __bf16* E     = (__bf16*)take((size_t)BB * KK * 2);       //  64 MB
    float*  rowsum= (float*) take((size_t)BB * 4);
    float*  lse   = (float*) take((size_t)BB * 4);
    float*  maskf = (float*) take((size_t)BB * 4);
    float*  U     = (float*) take((size_t)BB * USTRIDE * 4);  //   2 MB

    k_init<<<(BB * USTRIDE + 255) / 256, 256, 0, stream>>>(out, U);
    k_convA<<<(M1 * DD) / 256, 256, 0, stream>>>(norm_q, k_feat, bfA);
    k_transQ<<<dim3(KK / 32, DD / 32), dim3(32, 32), 0, stream>>>(queue, bfQT);
    k_convQLP<<<8192, 256, 0, stream>>>(qlp, bfQLP);
    k_gemm1<<<dim3(KK / 128, M1 / 128), 256, 0, stream>>>(bfA, bfQT, S);
    k_expdc<<<BB, 256, 0, stream>>>(S, E, rowsum);
    k_topk<<<BB, 256, 0, stream>>>(S, qlabel, target, out);
    k_gemm2<<<dim3((CC + 127) / 128, BB / 128, 32), 256, 0, stream>>>(E, bfQLP, U);
    k_fc<<<BB, 256, 0, stream>>>(q_logits, target, out, lse, maskf);
    k_dc<<<BB, 256, 0, stream>>>(U, rowsum, q_logits, lse, maskf, out);
}
